// GNN_node_70188355551331
// MI455X (gfx1250) — matrix operand order repaired and checked by reading, not compiled, not run on hardware
//
#include <hip/hip_runtime.h>

// ---------------- problem sizes ----------------
#define NN      300000
#define MM      200000
#define ESRC    300000
#define ESINK   1500000
#define EMB     128

// ---------------- vector types -----------------
typedef __bf16 bf16_t;
typedef __attribute__((ext_vector_type(16))) __bf16 v16bf;
typedef __attribute__((ext_vector_type(8)))  float  v8f;
typedef __attribute__((ext_vector_type(4)))  float  v4f;

enum { ACT_NONE = 0, ACT_RELU = 1, ACT_LRELU = 2, ACT_ABS = 3 };

// =====================================================================
// Fused WMMA GEMM:  Y = act( [ADDY ? Y : 0] + X[rows,K] @ W[K,N] + b[N] )
//   K, N, ACT, ADDY are compile-time: all tile guards fold away, the
//   kt/nt loops fully unroll, and each k-step is
//       {2*NT x ds_load_b128} -> one wait -> {NT x v_wmma} back-to-back.
//   256 threads = 8 waves; workgroup covers 128 rows (16 rows/wave).
//   Requires rows % 16 == 0 (true: 300000 / 200000).
// =====================================================================
template <int K, int N, int ACT, int ADDY>
__global__ void __launch_bounds__(256)
wmma_gemm(const float* __restrict__ X,
          const float* __restrict__ W,
          const float* __restrict__ Bias,
          float* __restrict__ Y,
          int rows)
{
    constexpr int Kpad = (K + 31) & ~31;
    constexpr int Nlds = (N + 15) & ~15;
    constexpr int NT   = Nlds >> 4;
    constexpr int KT   = Kpad >> 5;

    __shared__ bf16_t Wlds[Nlds * Kpad];   // <= 32 KB

    const int tid = threadIdx.x;
    // cooperative f32->bf16 staging of W, column-major [n][k], zero-padded
    for (int i = tid; i < Nlds * Kpad; i += 256) {
        const int n = i / Kpad, k = i - n * Kpad;
        float w = 0.0f;
        if ((k < K) && (Nlds == N || n < N)) w = W[k * N + n];
        Wlds[i] = static_cast<bf16_t>(w);
    }
    __syncthreads();

    const int wave    = tid >> 5;
    const int lane    = tid & 31;
    const int rowbase = blockIdx.x * 128 + wave * 16;
    if (rowbase >= rows) return;          // wave-uniform: EXEC stays all-ones

    const int hi  = lane >> 4;            // half-wave select
    const int lm  = lane & 15;
    // A layout (matches B): lanes 0-15 hold K {0..15}, lanes 16-31 {16..31}
    const int kA0 = hi ? 16 : 0;
    const int kA1 = hi ? 24 : 8;

    v8f acc[NT];
    #pragma unroll
    for (int t = 0; t < NT; ++t)
        acc[t] = v8f{0.f, 0.f, 0.f, 0.f, 0.f, 0.f, 0.f, 0.f};

    const float* xrow = X + (long long)(rowbase + lm) * K;

    #pragma unroll
    for (int kt = 0; kt < KT; ++kt) {
        const int k0 = kt * 32;

        // ---- A fragment: 4 guarded float4 chunks, converted to bf16 ----
        union { v16bf v; bf16_t e[16]; } a;
        const int offs[4] = { k0 + kA0, k0 + kA0 + 4, k0 + kA1, k0 + kA1 + 4 };
        #pragma unroll
        for (int c = 0; c < 4; ++c) {
            union { v4f v; float e[4]; } f;
            if (K == Kpad || offs[c] < K) f.v = *(const v4f*)(xrow + offs[c]);
            else { f.e[0] = 0.f; f.e[1] = 0.f; f.e[2] = 0.f; f.e[3] = 0.f; }
            #pragma unroll
            for (int j = 0; j < 4; ++j) a.e[c * 4 + j] = static_cast<bf16_t>(f.e[j]);
        }

        // ---- all B fragments first (batches the ds loads / single wait) ----
        const int kb = k0 + hi * 16;      // contiguous k = 16*hi + s, same order as A
        v16bf b[NT];
        #pragma unroll
        for (int t = 0; t < NT; ++t)
            b[t] = *(const v16bf*)(&Wlds[(t * 16 + lm) * Kpad + kb]);

        // ---- NT back-to-back WMMAs (independent accumulators: no hazard) ----
        #pragma unroll
        for (int t = 0; t < NT; ++t)
            acc[t] = __builtin_amdgcn_wmma_f32_16x16x32_bf16(
                false, a.v, false, b[t], (short)0, acc[t], false, false);
    }

    // ---- epilogue: bias + optional residual + activation (branch-free) ----
    // C/D layout: VGPR v, lane -> row = rowbase + 8*hi + v, col = t*16 + lm
    #pragma unroll
    for (int t = 0; t < NT; ++t) {
        const int n = t * 16 + lm;
        if (Nlds == N || n < N) {
            const float bia = Bias[n];
            union { v8f v; float e[8]; } u; u.v = acc[t];
            #pragma unroll
            for (int v = 0; v < 8; ++v) {
                const long long r   = rowbase + hi * 8 + v;
                const long long idx = r * (long long)N + n;
                float val = u.e[v] + bia;
                if (ADDY) val += Y[idx];
                if      (ACT == ACT_RELU)  val = val > 0.f ? val : 0.f;
                else if (ACT == ACT_LRELU) val = val > 0.f ? val : 0.01f * val;
                else if (ACT == ACT_ABS)   val = fabsf(val);
                Y[idx] = val;
            }
        }
    }
}

template <int K, int N, int ACT, int ADDY>
static inline void launch_gemm(const float* X, const float* W, const float* B,
                               float* Y, int rows, hipStream_t s) {
    wmma_gemm<K, N, ACT, ADDY><<<(rows + 127) / 128, 256, 0, s>>>(X, W, B, Y, rows);
}

// ------------------------ sparse / elementwise kernels ------------------------
__global__ void zero_f32(float* __restrict__ p, long long n) {
    long long i = (long long)blockIdx.x * blockDim.x + threadIdx.x;
    if (i < n) p[i] = 0.0f;
}

__global__ void lrelu_inplace(float* __restrict__ p, long long n) {
    long long i = (long long)blockIdx.x * blockDim.x + threadIdx.x;
    if (i < n) { float v = p[i]; p[i] = v > 0.f ? v : 0.01f * v; }
}

// agg[net] += h[node]; thread = (edge, 4 dims) so edge metadata loads amortize
__global__ void scatter_src(const float* __restrict__ h,
                            const int* __restrict__ node_idx,
                            const int* __restrict__ net_idx,
                            float* __restrict__ agg, long long total /* E*32 */) {
    long long t = (long long)blockIdx.x * blockDim.x + threadIdx.x;
    if (t >= total) return;
    const int e  = (int)(t >> 5);
    const int d0 = (int)(t & 31) * 4;
    const float* src = h   + (long long)node_idx[e] * EMB + d0;
    float*       dst = agg + (long long)net_idx[e]  * EMB + d0;
    const v4f v = *(const v4f*)src;
    atomicAdd(dst + 0, v.x); atomicAdd(dst + 1, v.y);
    atomicAdd(dst + 2, v.z); atomicAdd(dst + 3, v.w);
}

// agg[node] += (h_net[net] + attr[e]*w_attr) * (keep[e] ? ew[e] : 0)
__global__ void scatter_sink(const float* __restrict__ h_net,
                             const int* __restrict__ node_idx,
                             const int* __restrict__ net_idx,
                             const float* __restrict__ ew,
                             const float* __restrict__ attr,
                             const unsigned char* __restrict__ keep,
                             const float* __restrict__ w_attr,
                             float* __restrict__ agg, long long total /* E*32 */) {
    long long t = (long long)blockIdx.x * blockDim.x + threadIdx.x;
    if (t >= total) return;
    const int e  = (int)(t >> 5);
    const int d0 = (int)(t & 31) * 4;
    const float w = keep[e] ? ew[e] : 0.0f;
    if (w == 0.0f) return;
    const float at = attr[e];
    const v4f hv = *(const v4f*)(h_net + (long long)net_idx[e] * EMB + d0);
    const v4f wa = *(const v4f*)(w_attr + d0);
    float*   dst = agg + (long long)node_idx[e] * EMB + d0;
    atomicAdd(dst + 0, (hv.x + at * wa.x) * w);
    atomicAdd(dst + 1, (hv.y + at * wa.y) * w);
    atomicAdd(dst + 2, (hv.z + at * wa.z) * w);
    atomicAdd(dst + 3, (hv.w + at * wa.w) * w);
}

// =====================================================================
extern "C" void kernel_launch(void* const* d_in, const int* in_sizes, int n_in,
                              void* d_out, int out_size, void* d_ws, size_t ws_size,
                              hipStream_t stream) {
    const float* node_x = (const float*)d_in[0];
    const float* net_x  = (const float*)d_in[1];
    const int*   ei_src  = (const int*)d_in[2];   // [2, ESRC]
    const int*   ei_sink = (const int*)d_in[3];   // [2, ESINK]
    const float* ew      = (const float*)d_in[4];
    const float* attr    = (const float*)d_in[5];
    const unsigned char* keep = (const unsigned char*)d_in[6];

    auto P = [&](int i) { return (const float*)d_in[i]; };
    // params flattened in insertion (dict) order
    const float *ne_w1 = P(7),  *ne_b1 = P(8),  *ne_w2 = P(9),  *ne_b2 = P(10);   // node_enc
    const float *me_w1 = P(11), *me_b1 = P(12), *me_w2 = P(13), *me_b2 = P(14);   // net_enc
    struct Layer { const float *pw1,*pb1,*pw2,*pb2,*sw1,*sb1,*sw2,*sb2,*wat; } L[3];
    for (int l = 0; l < 3; ++l) {
        int b = 15 + 9 * l;
        L[l] = { P(b), P(b+1), P(b+2), P(b+3), P(b+4), P(b+5), P(b+6), P(b+7), P(b+8) };
    }
    const float *f1n_w = P(42), *f1n_b = P(43), *f2n_w = P(44), *f2n_b = P(45);
    const float *f1m_w = P(46), *f1m_b = P(47), *f2m_w = P(48), *f2m_b = P(49);

    // workspace layout (floats)
    float* ws    = (float*)d_ws;
    float* h     = ws;                          // N * 128
    float* h_net = h     + (long long)NN * EMB; // M * 128
    float* T     = h_net + (long long)MM * EMB; // N * 128 (temp hidden)
    float* agg   = T     + (long long)NN * EMB; // N * 128 (shared src/sink agg)
    float* out_node = (float*)d_out;            // [N, 8]
    float* out_net  = out_node + (long long)NN * 8; // [M, 8]

    // encoders: h = lrelu(x@w1+b1) @ w2 + b2
    launch_gemm<32, 128, ACT_LRELU, 0>(node_x, ne_w1, ne_b1, T,     NN, stream);
    launch_gemm<128,128, ACT_NONE,  0>(T,      ne_w2, ne_b2, h,     NN, stream);
    launch_gemm<16, 128, ACT_LRELU, 0>(net_x,  me_w1, me_b1, T,     MM, stream);
    launch_gemm<128,128, ACT_NONE,  0>(T,      me_w2, me_b2, h_net, MM, stream);

    const long long nM = (long long)MM * EMB, nN = (long long)NN * EMB;
    const long long te = (long long)ESRC * 32, ts = (long long)ESINK * 32;

    for (int l = 0; l < 3; ++l) {
        // (1) driver nodes -> nets
        zero_f32<<<(int)((nM + 255) / 256), 256, 0, stream>>>(agg, nM);
        scatter_src<<<(int)((te + 255) / 256), 256, 0, stream>>>(
            h, ei_src, ei_src + ESRC, agg, te);
        launch_gemm<128,128, ACT_RELU, 0>(agg, L[l].pw1, L[l].pb1, T,     MM, stream);
        launch_gemm<128,128, ACT_NONE, 1>(T,   L[l].pw2, L[l].pb2, h_net, MM, stream);
        // (2) nets -> sink nodes (uses pre-lrelu h_net)
        zero_f32<<<(int)((nN + 255) / 256), 256, 0, stream>>>(agg, nN);
        scatter_sink<<<(int)((ts + 255) / 256), 256, 0, stream>>>(
            h_net, ei_sink, ei_sink + ESINK, ew, attr, keep, L[l].wat, agg, ts);
        lrelu_inplace<<<(int)((nM + 255) / 256), 256, 0, stream>>>(h_net, nM);
        launch_gemm<128,128, ACT_RELU,  0>(agg, L[l].sw1, L[l].sb1, T, NN, stream);
        launch_gemm<128,128, ACT_LRELU, 1>(T,   L[l].sw2, L[l].sb2, h, NN, stream);
    }

    // heads
    launch_gemm<128,128, ACT_LRELU, 0>(h,     f1n_w, f1n_b, T,        NN, stream);
    launch_gemm<128,  8, ACT_ABS,   0>(T,     f2n_w, f2n_b, out_node, NN, stream);
    launch_gemm<128, 64, ACT_LRELU, 0>(h_net, f1m_w, f1m_b, T,        MM, stream);
    launch_gemm<64,   8, ACT_ABS,   0>(T,     f2m_w, f2m_b, out_net,  MM, stream);
}
